// CGCLayer_4647154614944
// MI455X (gfx1250) — matrix operand order repaired and checked by reading, not compiled, not run on hardware
//
#include <hip/hip_runtime.h>
#include <hip/hip_bf16.h>
#include <type_traits>

typedef __attribute__((ext_vector_type(16))) __bf16 v16bf;
typedef __attribute__((ext_vector_type(8)))  __bf16 v8bf;
typedef __attribute__((ext_vector_type(8)))  float  v8f;

#define D_MODEL    1024
#define EXPERT_DIM 1024
#define BATCH      8192
#define NTASK      3
#define NCAND      8

#define ROWS 128
#define COLS 64
#define KC   32
#define ASTR 40   // LDS row stride in halfs (80B: keeps 16B alignment, staggers banks)
#define BSTR 40

// ---------------------------------------------------------------------------
// Kernel 1: gate logits + softmax.  One wave per row, lanes split the d-dim.
// ---------------------------------------------------------------------------
__global__ __launch_bounds__(256) void cgc_gates_kernel(
    const float* __restrict__ x, const float* __restrict__ Wg,
    float* __restrict__ gates)
{
  const int lane = threadIdx.x & 31;
  const int wid  = threadIdx.x >> 5;
  const int row  = blockIdx.x * 8 + wid;

  float acc[24];
#pragma unroll
  for (int i = 0; i < 24; i++) acc[i] = 0.f;

  const float* xr = x + (size_t)row * D_MODEL;
  for (int d = lane; d < D_MODEL; d += 32) {
    float xv = xr[d];
#pragma unroll
    for (int t = 0; t < NTASK; t++) {
      const float4* w4 = (const float4*)(Wg + ((size_t)t * D_MODEL + d) * 8);
      float4 a = w4[0], b = w4[1];
      acc[t*8+0] += xv*a.x; acc[t*8+1] += xv*a.y;
      acc[t*8+2] += xv*a.z; acc[t*8+3] += xv*a.w;
      acc[t*8+4] += xv*b.x; acc[t*8+5] += xv*b.y;
      acc[t*8+6] += xv*b.z; acc[t*8+7] += xv*b.w;
    }
  }
#pragma unroll
  for (int i = 0; i < 24; i++) {
#pragma unroll
    for (int m = 16; m >= 1; m >>= 1) acc[i] += __shfl_xor(acc[i], m, 32);
  }
  if (lane < 24) {
    int t = lane >> 3, e = lane & 7;
    float mx = acc[t*8];
#pragma unroll
    for (int k = 1; k < 8; k++) mx = fmaxf(mx, acc[t*8+k]);
    float den = 0.f;
#pragma unroll
    for (int k = 0; k < 8; k++) den += __expf(acc[t*8+k] - mx);
    gates[((size_t)t * BATCH + row) * NCAND + e] = __expf(acc[lane] - mx) / den;
  }
}

// ---------------------------------------------------------------------------
// Kernel 2: fused experts + bias + ReLU + gated mixture.
// Block 256 thr (8 waves as 4 row x 2 col), tile 128x64, 3 task accumulators.
// Split-bf16 (hi+lo): Ah*Bh + Ah*Bl + Al*Bh  -> ~fp32 accuracy on WMMA.
// All oacc indices are compile-time constants (no scratch spilling).
// ---------------------------------------------------------------------------
__global__ __launch_bounds__(256) void cgc_expert_kernel(
    const float* __restrict__ x,
    const float* __restrict__ Ws, const float* __restrict__ bs,
    const float* __restrict__ Wt, const float* __restrict__ bt,
    const float* __restrict__ gates,
    float* __restrict__ out)
{
  __shared__ __bf16 Ah[ROWS][ASTR];
  __shared__ __bf16 Al[ROWS][ASTR];
  __shared__ __bf16 Bh[COLS][BSTR];
  __shared__ __bf16 Bl[COLS][BSTR];
  __shared__ float  gl[NTASK][ROWS][NCAND];

  const int tid  = threadIdx.x;
  const int lane = tid & 31;
  const int wid  = tid >> 5;
  const int wrow = (wid & 3) * 32;   // wave local row base
  const int wcol = (wid >> 2) * 32;  // wave local col base
  const int row0 = blockIdx.y * ROWS;
  const int col0 = blockIdx.x * COLS;

  // stage softmaxed gates for this row tile (visible after first barrier)
  for (int j = tid; j < NTASK * ROWS * NCAND; j += 256) {
    int t = j / (ROWS * NCAND);
    int r = (j / NCAND) % ROWS;
    int e = j & (NCAND - 1);
    gl[t][r][e] = gates[((size_t)t * BATCH + row0 + r) * NCAND + e];
  }

  const int mrow  = lane & 15;
  const int g0    = (lane < 16) ? 0 : 8;   // A frag K-group base (ISA layout)
  const int kb    = (lane < 16) ? 0 : 16;  // unused; B frags use g0, same K order as A
  const int half8 = (lane >> 4) << 3;      // C/D: lanes 16-31 hold M+8

  v8f oacc[NTASK][2][2];
#pragma unroll
  for (int t = 0; t < NTASK; t++)
#pragma unroll
    for (int a = 0; a < 2; a++)
#pragma unroll
      for (int b = 0; b < 2; b++)
#pragma unroll
        for (int i = 0; i < 8; i++) oacc[t][a][b][i] = 0.f;

  // ---- one expert GEMM over the full K dim into acc[2][2] ----
  auto run_gemm = [&](const float* __restrict__ W, v8f (&acc)[2][2]) {
#pragma unroll
    for (int a = 0; a < 2; a++)
#pragma unroll
      for (int b = 0; b < 2; b++)
#pragma unroll
        for (int i = 0; i < 8; i++) acc[a][b][i] = 0.f;

    for (int ko = 0; ko < D_MODEL; ko += KC) {
      __syncthreads();  // previous iteration's fragment reads done
      // ---- stage A chunk: 128x32 f32 -> hi/lo bf16, packed b64 stores ----
#pragma unroll
      for (int j = 0; j < 4; j++) {
        int li = tid + j * 256;            // float4 index, 0..1023
        int r  = li >> 3;
        int c4 = li & 7;
        const float* gp = x + (size_t)(row0 + r) * D_MODEL + ko + c4 * 4;
        float4 v = *(const float4*)gp;
        __builtin_prefetch(gp + KC, 0, 3);       // next K-chunk of x
        float vv[4] = {v.x, v.y, v.z, v.w};
        union { __bf16 h[4]; uint2 u; } ph, pl;
#pragma unroll
        for (int q = 0; q < 4; q++) {
          __bf16 h = (__bf16)vv[q];
          ph.h[q] = h;
          pl.h[q] = (__bf16)(vv[q] - (float)h);
        }
        *(uint2*)&Ah[r][c4 * 4] = ph.u;          // 8B-aligned ds_store_b64
        *(uint2*)&Al[r][c4 * 4] = pl.u;
      }
      // ---- stage B chunk transposed: 32x64 f32 -> hi/lo bf16 [n][k] ----
#pragma unroll
      for (int j = 0; j < 2; j++) {
        int li = tid + j * 256;            // float4 index, 0..511
        int kr = li >> 4;
        int c4 = li & 15;
        const float* gp = W + (size_t)(ko + kr) * EXPERT_DIM + col0 + c4 * 4;
        float4 v = *(const float4*)gp;
        __builtin_prefetch(gp + KC * EXPERT_DIM, 0, 3);  // next K-chunk of W
        float vv[4] = {v.x, v.y, v.z, v.w};
#pragma unroll
        for (int q = 0; q < 4; q++) {
          __bf16 h = (__bf16)vv[q];
          int n = c4 * 4 + q;
          Bh[n][kr] = h;
          Bl[n][kr] = (__bf16)(vv[q] - (float)h);
        }
      }
      __syncthreads();

      // ---- A fragments (16x32 bf16 layout per ISA §7.12.2) ----
      v16bf ahf[2], alf[2];
#pragma unroll
      for (int fr = 0; fr < 2; fr++) {
        int m = wrow + fr * 16 + mrow;
        v8bf p0 = *(const v8bf*)&Ah[m][g0];
        v8bf p1 = *(const v8bf*)&Ah[m][g0 + 16];
        v8bf q0 = *(const v8bf*)&Al[m][g0];
        v8bf q1 = *(const v8bf*)&Al[m][g0 + 16];
#pragma unroll
        for (int q = 0; q < 8; q++) {
          ahf[fr][q] = p0[q]; ahf[fr][8 + q] = p1[q];
          alf[fr][q] = q0[q]; alf[fr][8 + q] = q1[q];
        }
      }
      // ---- B fragments + 12 WMMAs ----
#pragma unroll
      for (int fc = 0; fc < 2; fc++) {
        int n = wcol + fc * 16 + mrow;
        v8bf p0 = *(const v8bf*)&Bh[n][g0];
        v8bf p1 = *(const v8bf*)&Bh[n][g0 + 16];
        v8bf q0 = *(const v8bf*)&Bl[n][g0];
        v8bf q1 = *(const v8bf*)&Bl[n][g0 + 16];
        v16bf bh, bl;
#pragma unroll
        for (int q = 0; q < 8; q++) {
          bh[q] = p0[q]; bh[8 + q] = p1[q];
          bl[q] = q0[q]; bl[8 + q] = q1[q];
        }
#pragma unroll
        for (int fr = 0; fr < 2; fr++) {
          acc[fr][fc] = __builtin_amdgcn_wmma_f32_16x16x32_bf16(
              false, ahf[fr], false, bh, (short)0, acc[fr][fc], false, false);
          acc[fr][fc] = __builtin_amdgcn_wmma_f32_16x16x32_bf16(
              false, ahf[fr], false, bl, (short)0, acc[fr][fc], false, false);
          acc[fr][fc] = __builtin_amdgcn_wmma_f32_16x16x32_bf16(
              false, alf[fr], false, bh, (short)0, acc[fr][fc], false, false);
        }
      }
    }
  };

  v8f acc[2][2];

  // ---- task-specific experts: T is a compile-time constant per call ----
  auto run_task = [&](auto TC) {
    constexpr int T = decltype(TC)::value;
    for (int j = 0; j < 4; j++) {   // expert within task (dynamic is fine)
      run_gemm(Wt + (size_t)(T * 4 + j) * D_MODEL * EXPERT_DIM, acc);
      const float* bias = bt + (T * 4 + j) * EXPERT_DIM;
#pragma unroll
      for (int fc = 0; fc < 2; fc++) {
        float bcol = bias[col0 + wcol + fc * 16 + mrow];
#pragma unroll
        for (int fr = 0; fr < 2; fr++)
#pragma unroll
          for (int i = 0; i < 8; i++) {
            float v = fmaxf(acc[fr][fc][i] + bcol, 0.f);
            int rl = wrow + fr * 16 + i + half8;
            oacc[T][fr][fc][i] += gl[T][rl][j] * v;   // constant indices
          }
      }
    }
  };
  run_task(std::integral_constant<int, 0>{});
  run_task(std::integral_constant<int, 1>{});
  run_task(std::integral_constant<int, 2>{});

  // ---- shared experts: feed all 3 tasks with constant task indices ----
  for (int s = 0; s < 4; s++) {
    run_gemm(Ws + (size_t)s * D_MODEL * EXPERT_DIM, acc);
    const float* bias = bs + s * EXPERT_DIM;
#pragma unroll
    for (int fc = 0; fc < 2; fc++) {
      float bcol = bias[col0 + wcol + fc * 16 + mrow];
#pragma unroll
      for (int fr = 0; fr < 2; fr++)
#pragma unroll
        for (int i = 0; i < 8; i++) {
          float v = fmaxf(acc[fr][fc][i] + bcol, 0.f);
          int rl = wrow + fr * 16 + i + half8;
#pragma unroll
          for (int t = 0; t < NTASK; t++)
            oacc[t][fr][fc][i] += gl[t][rl][4 + s] * v;
        }
    }
  }

  // ---- write (3, B, H) output ----
#pragma unroll
  for (int t = 0; t < NTASK; t++)
#pragma unroll
    for (int fr = 0; fr < 2; fr++)
#pragma unroll
      for (int fc = 0; fc < 2; fc++)
#pragma unroll
        for (int i = 0; i < 8; i++) {
          int r = row0 + wrow + fr * 16 + i + half8;
          int c = col0 + wcol + fc * 16 + mrow;
          out[((size_t)t * BATCH + r) * EXPERT_DIM + c] = oacc[t][fr][fc][i];
        }
}

extern "C" void kernel_launch(void* const* d_in, const int* in_sizes, int n_in,
                              void* d_out, int out_size, void* d_ws, size_t ws_size,
                              hipStream_t stream) {
  const float* x  = (const float*)d_in[0];
  const float* Ws = (const float*)d_in[1];
  const float* bs = (const float*)d_in[2];
  const float* Wt = (const float*)d_in[3];
  const float* bt = (const float*)d_in[4];
  const float* Wg = (const float*)d_in[5];
  float* out   = (float*)d_out;
  float* gates = (float*)d_ws;  // 3*8192*8 f32 = 768 KB scratch, fully written by kernel 1

  cgc_gates_kernel<<<BATCH / 8, 256, 0, stream>>>(x, Wg, gates);
  cgc_expert_kernel<<<dim3(EXPERT_DIM / COLS, BATCH / ROWS), 256, 0, stream>>>(
      x, Ws, bs, Wt, bt, gates, out);
}
